// Aggregator_22548578304241
// MI455X (gfx1250) — compile-verified, matrix operand order checked by reading, not run on hardware
//
#include <hip/hip_runtime.h>

typedef __attribute__((ext_vector_type(4)))  float    v4f;
typedef __attribute__((ext_vector_type(8)))  float    v8f;
typedef __attribute__((ext_vector_type(4)))  _Float16 v4h;
typedef __attribute__((ext_vector_type(8)))  _Float16 v8h;
typedef __attribute__((ext_vector_type(16))) _Float16 v16h;

#define FD      256          // feature dim D
#define TWO_D   512          // concat dim
#define OUT_D   256          // output dim
#define KNB     16           // neighbors per node
#define ROWS    32           // rows (nodes) per block
#define LDP     (TWO_D + 8)  // LDS row pitch in halfs (+16B pad -> conflict-free)

__device__ __forceinline__ v16h join16(v8h lo, v8h hi) {
    v16h r;
    #pragma unroll
    for (int i = 0; i < 8; ++i) { r[i] = lo[i]; r[i + 8] = hi[i]; }
    return r;
}

__device__ __forceinline__ v16h cvt16(const float* __restrict__ p) {
    v4f c0 = *(const v4f*)(p);
    v4f c1 = *(const v4f*)(p + 4);
    v4f c2 = *(const v4f*)(p + 16);
    v4f c3 = *(const v4f*)(p + 20);
    v16h r;
    #pragma unroll
    for (int i = 0; i < 4; ++i) {
        r[i]      = (_Float16)c0[i];
        r[i + 4]  = (_Float16)c1[i];
        r[i + 8]  = (_Float16)c2[i];
        r[i + 12] = (_Float16)c3[i];
    }
    return r;
}

// One-time fp32 -> f16 conversion of W into workspace.
__global__ __launch_bounds__(256) void wcvt_kernel(const float* __restrict__ W,
                                                   _Float16* __restrict__ Wh, int n)
{
    int i = (blockIdx.x * 256 + threadIdx.x) * 4;
    if (i < n) {
        v4f w = *(const v4f*)(W + i);
        v4h o;
        #pragma unroll
        for (int k = 0; k < 4; ++k) o[k] = (_Float16)w[k];
        *(v4h*)(Wh + i) = o;
    }
}

template <bool USE_WH>
__global__ __launch_bounds__(256) void sage_fused_wmma(
    const float*    __restrict__ h,
    const float*    __restrict__ nei,
    const float*    __restrict__ W,     // fp32 weights (fallback path)
    const _Float16* __restrict__ Wh,    // f16 weights in d_ws (fast path)
    const float*    __restrict__ bias,
    float*          __restrict__ out,
    int N)
{
    __shared__ _Float16 cat[ROWS * LDP];   // 32 rows x 520 halfs = 33.3 KB

    const int tid      = threadIdx.x;
    const int row_base = blockIdx.x * ROWS;

    // ---------------- Phase 1: cat = [h | mean(nei)] in LDS (f16) ----------------
    {
        // Warm the weight stream (L2-resident) via the CDNA5 prefetch path.
        #pragma unroll
        for (int p = 0; p < 4; ++p) {
            if constexpr (USE_WH)
                __builtin_prefetch(Wh + ((size_t)(tid + p * 256) * 128), 0, 3);
            else
                __builtin_prefetch(W + ((size_t)(tid + p * 256) * 128), 0, 3);
        }

        const int r4 = tid >> 6;          // 0..3
        const int d  = (tid & 63) << 2;   // 0..252 step 4 (float4 lanes)

        #pragma unroll
        for (int rr = 0; rr < ROWS / 4; ++rr) {
            int r   = (rr << 2) + r4;               // 0..31
            int row = row_base + r;
            if (row >= N) row = N - 1;              // tail clamp (loads only)

            const v4f hv = *(const v4f*)(h + (size_t)row * FD + d);

            const float* np = nei + ((size_t)row * KNB) * FD + d;
            v4f s = (v4f)0.0f;
            #pragma unroll
            for (int j = 0; j < KNB; ++j)
                s += *(const v4f*)(np + (size_t)j * FD);
            s *= 0.0625f;   // exact fp32 mean over 16 neighbors

            v4h hh, sh;
            #pragma unroll
            for (int i = 0; i < 4; ++i) {
                hh[i] = (_Float16)hv[i];
                sh[i] = (_Float16)s[i];
            }
            *(v4h*)(&cat[r * LDP + d])      = hh;
            *(v4h*)(&cat[r * LDP + FD + d]) = sh;
        }
    }
    __syncthreads();

    // -------- Phase 2: [32 x 512] x [512 x 256] GEMM, 2x2 WMMA tile per wave --------
    const int wave    = tid >> 5;        // 0..7 -> N-tile pair
    const int lane    = tid & 31;
    const int halfsel = lane >> 4;       // K-half select (ISA 16-bit A/B layout)
    const int mlo     = lane & 15;

    const int ob0 = wave * 32;           // two 16-wide output column tiles
    const int ob1 = ob0 + 16;

    const _Float16* arow0 = &cat[mlo * LDP];          // M-tile 0 (rows 0..15)
    const _Float16* arow1 = &cat[(16 + mlo) * LDP];   // M-tile 1 (rows 16..31)

    const _Float16* wh0 = Wh + (size_t)(ob0 + mlo) * TWO_D;
    const _Float16* wh1 = Wh + (size_t)(ob1 + mlo) * TWO_D;
    const float*    wf0 = W  + (size_t)(ob0 + mlo) * TWO_D;
    const float*    wf1 = W  + (size_t)(ob1 + mlo) * TWO_D;

    v8f acc00 = {};   // (M-tile0, N-tile0)
    v8f acc01 = {};   // (M-tile0, N-tile1)
    v8f acc10 = {};   // (M-tile1, N-tile0)
    v8f acc11 = {};   // (M-tile1, N-tile1)

    #pragma unroll 4
    for (int ks = 0; ks < TWO_D / 32; ++ks) {
        const int kb = ks * 32 + halfsel * 8;  // chunk0: K kb..kb+7, chunk1: +16

        v16h a0 = join16(*(const v8h*)(arow0 + kb), *(const v8h*)(arow0 + kb + 16));
        v16h a1 = join16(*(const v8h*)(arow1 + kb), *(const v8h*)(arow1 + kb + 16));

        v16h b0, b1;
        if constexpr (USE_WH) {
            b0 = join16(*(const v8h*)(wh0 + kb), *(const v8h*)(wh0 + kb + 16));
            b1 = join16(*(const v8h*)(wh1 + kb), *(const v8h*)(wh1 + kb + 16));
        } else {
            b0 = cvt16(wf0 + kb);
            b1 = cvt16(wf1 + kb);
        }

        acc00 = __builtin_amdgcn_wmma_f32_16x16x32_f16(false, a0, false, b0, (short)0, acc00, false, false);
        acc10 = __builtin_amdgcn_wmma_f32_16x16x32_f16(false, a1, false, b0, (short)0, acc10, false, false);
        acc01 = __builtin_amdgcn_wmma_f32_16x16x32_f16(false, a0, false, b1, (short)0, acc01, false, false);
        acc11 = __builtin_amdgcn_wmma_f32_16x16x32_f16(false, a1, false, b1, (short)0, acc11, false, false);
    }

    // -------- Epilogue: bias + relu + store (C/D layout: M = v + 8*halfsel, N = mlo) --------
    const float bv0 = bias[ob0 + mlo];
    const float bv1 = bias[ob1 + mlo];

    #pragma unroll
    for (int v = 0; v < 8; ++v) {
        const int m    = v + (halfsel << 3);
        const int row0 = row_base + m;        // M-tile 0
        const int row1 = row_base + 16 + m;   // M-tile 1
        if (row0 < N) {
            float r0 = acc00[v] + bv0; r0 = r0 > 0.0f ? r0 : 0.0f;
            float r1 = acc01[v] + bv1; r1 = r1 > 0.0f ? r1 : 0.0f;
            out[(size_t)row0 * OUT_D + ob0 + mlo] = r0;
            out[(size_t)row0 * OUT_D + ob1 + mlo] = r1;
        }
        if (row1 < N) {
            float r0 = acc10[v] + bv0; r0 = r0 > 0.0f ? r0 : 0.0f;
            float r1 = acc11[v] + bv1; r1 = r1 > 0.0f ? r1 : 0.0f;
            out[(size_t)row1 * OUT_D + ob0 + mlo] = r0;
            out[(size_t)row1 * OUT_D + ob1 + mlo] = r1;
        }
    }
}

extern "C" void kernel_launch(void* const* d_in, const int* in_sizes, int n_in,
                              void* d_out, int out_size, void* d_ws, size_t ws_size,
                              hipStream_t stream)
{
    const float* h   = (const float*)d_in[0];   // [N, 256]
    const float* nei = (const float*)d_in[1];   // [N, 16, 256]
    const float* W   = (const float*)d_in[2];   // [256, 512]
    const float* b   = (const float*)d_in[3];   // [256]
    float*       out = (float*)d_out;           // [N, 256]

    const int N      = in_sizes[0] / FD;
    const int blocks = (N + ROWS - 1) / ROWS;

    const int    wn      = TWO_D * OUT_D;                       // 131072 weights
    const size_t wh_need = (size_t)wn * sizeof(_Float16);       // 256 KB

    if (ws_size >= wh_need) {
        _Float16* Wh = (_Float16*)d_ws;
        wcvt_kernel<<<wn / (256 * 4), 256, 0, stream>>>(W, Wh, wn);
        sage_fused_wmma<true><<<blocks, 256, 0, stream>>>(h, nei, W, Wh, b, out, N);
    } else {
        sage_fused_wmma<false><<<blocks, 256, 0, stream>>>(h, nei, W, nullptr, b, out, N);
    }
}
